// SpikingNeuralNetwork_83820581749444
// MI455X (gfx1250) — compile-verified, matrix operand order checked by reading, not run on hardware
//
#include <hip/hip_runtime.h>

typedef __attribute__((ext_vector_type(16))) _Float16 v16h;
typedef __attribute__((ext_vector_type(8)))  _Float16 v8h;
typedef __attribute__((ext_vector_type(8)))  float    v8f;

#define BB 64
#define TT 256
#define II 2048
#define HH 4096
#define OO 1024

#define THRESHOLD 1.0f
#define DECAY 0.9f

// ---------------------------------------------------------------------------
// Kernel 1: batch-mean over B, convert to f16.  xbar[t,i] = mean_b x[b,t,i]
// 134 MB streamed read -> 1 MB f16 write.  Coalesced over i.
// ---------------------------------------------------------------------------
__global__ void snn_batch_mean(const float* __restrict__ x,
                               _Float16* __restrict__ xbar) {
  int idx = blockIdx.x * blockDim.x + threadIdx.x;   // over T*I
  if (idx >= TT * II) return;
  const float* p = x + idx;
  float s = 0.f;
#pragma unroll 4
  for (int b = 0; b < BB; ++b) s += p[(long)b * (TT * II)];
  xbar[idx] = (_Float16)(s * (1.0f / BB));
}

// ---------------------------------------------------------------------------
// Kernel 2: transpose + convert W_in [I,H] f32 -> Wt [H,I] f16 via LDS tile.
// Makes the WMMA B-fragment a contiguous 16-byte load per lane.
// ---------------------------------------------------------------------------
__global__ void snn_transpose_w(const float* __restrict__ Win,
                                _Float16* __restrict__ Wt) {
  __shared__ _Float16 tile[32][33];
  const int i0 = blockIdx.y * 32;   // I dim
  const int h0 = blockIdx.x * 32;   // H dim
  const int tx = threadIdx.x;       // 0..31
  const int ty = threadIdx.y;       // 0..7
  for (int r = ty; r < 32; r += 8)
    tile[r][tx] = (_Float16)Win[(long)(i0 + r) * HH + h0 + tx];
  __syncthreads();
  for (int r = ty; r < 32; r += 8)
    Wt[(long)(h0 + r) * II + i0 + tx] = tile[tx][r];
}

// ---------------------------------------------------------------------------
// Kernel 3: currents[T,H] = xbar[T,I] @ W_in[I,H] + b_in  via WMMA f16->f32.
// One wave per 16x16 tile; K-loop of 64 x v_wmma_f32_16x16x32_f16.
// A-frag: lane L -> row L&15, two 16B loads at K-groups (L>>4)*8 and 16+...
// B-frag: lane L -> col L&15, same K grouping from N-major Wt.
// ---------------------------------------------------------------------------
__global__ void snn_gemm_wmma(const _Float16* __restrict__ A,    // [T,I]
                              const _Float16* __restrict__ Bt,   // [H,I]
                              const float*    __restrict__ bias, // [H]
                              float*          __restrict__ Cur)  // [T,H]
{
  const int lane = threadIdx.x;                 // 0..31
  const int wave = threadIdx.y;                 // 0..7
  const int m0 = blockIdx.y * 16;
  const int n0 = (blockIdx.x * 8 + wave) * 16;

  const int rc   = lane & 15;        // A row / B column owned by this lane
  const int kgrp = (lane >> 4) * 8;  // 0 or 8

  const _Float16* aRow = A  + (long)(m0 + rc) * II + kgrp;
  const _Float16* bCol = Bt + (long)(n0 + rc) * II + kgrp;

  v8f acc = {};
  for (int k = 0; k < II; k += 32) {
    union { v16h v; v8h h[2]; } af, bf;
    af.h[0] = *(const v8h*)(aRow + k);
    af.h[1] = *(const v8h*)(aRow + k + 16);
    bf.h[0] = *(const v8h*)(bCol + k);
    bf.h[1] = *(const v8h*)(bCol + k + 16);
    __builtin_prefetch(aRow + k + 128, 0, 1);   // global_prefetch_b8
    __builtin_prefetch(bCol + k + 128, 0, 1);
    // (neg_a, A, neg_b, B, c_mod, C, reuse_a, reuse_b)
    acc = __builtin_amdgcn_wmma_f32_16x16x32_f16(
        false, af.v, false, bf.v, (short)0, acc, false, false);
  }

  // C/D layout: lane L owns col L&15; VGPR r -> row r (+8 for lanes 16..31)
  const int nl  = lane & 15;
  const int mhi = (lane >> 4) * 8;
  const float bv = bias[n0 + nl];
#pragma unroll
  for (int r = 0; r < 8; ++r)
    Cur[(long)(m0 + mhi + r) * HH + n0 + nl] = acc[r] + bv;
}

// ---------------------------------------------------------------------------
// Kernel 4: sequential LIF scan over T, per neuron h.  Coalesced over h.
// ---------------------------------------------------------------------------
__global__ void snn_scan(const float* __restrict__ Cur,  // [T,H]
                         const float* __restrict__ m0v,  // [H]
                         float* __restrict__ agg) {      // [H]
  int h = blockIdx.x * blockDim.x + threadIdx.x;
  if (h >= HH) return;
  float m = m0v[h];
  float cnt = 0.f;
  for (int t = 0; t < TT; ++t) {
    m = DECAY * m + Cur[(long)t * HH + h];
    if (m > THRESHOLD) { cnt += 1.0f; m = 0.f; }
  }
  agg[h] = cnt * (1.0f / TT);
}

// ---------------------------------------------------------------------------
// Kernel 5: out[O] = agg[H] @ W_out[H,O] + b_out.  Coalesced over o.
// ---------------------------------------------------------------------------
__global__ void snn_out(const float* __restrict__ agg,
                        const float* __restrict__ Wout,   // [H,O]
                        const float* __restrict__ bout,
                        float* __restrict__ out) {
  int o = blockIdx.x * blockDim.x + threadIdx.x;
  if (o >= OO) return;
  float s = bout[o];
  for (int h = 0; h < HH; ++h)
    s += agg[h] * Wout[(long)h * OO + o];
  out[o] = s;
}

// ---------------------------------------------------------------------------
extern "C" void kernel_launch(void* const* d_in, const int* in_sizes, int n_in,
                              void* d_out, int out_size, void* d_ws, size_t ws_size,
                              hipStream_t stream) {
  const float* x     = (const float*)d_in[0];  // [B,T,I]
  const float* W_in  = (const float*)d_in[1];  // [I,H]
  const float* b_in  = (const float*)d_in[2];  // [H]
  const float* W_out = (const float*)d_in[3];  // [H,O]
  const float* b_out = (const float*)d_in[4];  // [O]
  const float* m0    = (const float*)d_in[5];  // [H]
  float* out = (float*)d_out;                  // [O]

  // Workspace layout (256B-aligned slices)
  char* ws = (char*)d_ws;
  size_t off = 0;
  _Float16* xbar = (_Float16*)(ws + off);           // T*I f16   = 1 MB
  off += (size_t)TT * II * sizeof(_Float16);
  off = (off + 255) & ~(size_t)255;
  _Float16* Wt = (_Float16*)(ws + off);             // H*I f16   = 16 MB
  off += (size_t)HH * II * sizeof(_Float16);
  off = (off + 255) & ~(size_t)255;
  float* Cur = (float*)(ws + off);                  // T*H f32   = 4 MB
  off += (size_t)TT * HH * sizeof(float);
  off = (off + 255) & ~(size_t)255;
  float* agg = (float*)(ws + off);                  // H f32
  off += (size_t)HH * sizeof(float);

  // 1) batch mean + f16 convert
  {
    int n = TT * II;
    snn_batch_mean<<<(n + 255) / 256, 256, 0, stream>>>(x, xbar);
  }
  // 2) W_in transpose/convert -> [H,I] f16
  {
    dim3 grid(HH / 32, II / 32);
    dim3 block(32, 8);
    snn_transpose_w<<<grid, block, 0, stream>>>(W_in, Wt);
  }
  // 3) WMMA GEMM -> currents [T,H]
  {
    dim3 grid(HH / 16 / 8, TT / 16);   // (32, 16)
    dim3 block(32, 8);                 // 8 waves, one 16x16 tile each
    snn_gemm_wmma<<<grid, block, 0, stream>>>(xbar, Wt, b_in, Cur);
  }
  // 4) LIF scan
  snn_scan<<<HH / 256, 256, 0, stream>>>(Cur, m0, agg);
  // 5) output GEMV
  snn_out<<<OO / 256, 256, 0, stream>>>(agg, W_out, b_out, out);
}
